// MemristorLinear_50826642981444
// MI455X (gfx1250) — matrix operand order repaired and checked by reading, not compiled, not run on hardware
//
#include <hip/hip_runtime.h>

typedef _Float16 v16h __attribute__((ext_vector_type(16)));
typedef float    v8f  __attribute__((ext_vector_type(8)));
typedef unsigned int v4u __attribute__((ext_vector_type(4)));
typedef int      v4i_gcc __attribute__((vector_size(16)));   // matches builtin param type

#define AS1 __attribute__((address_space(1)))
#define AS3 __attribute__((address_space(3)))

#define TOKENS   8192
#define D_IN     4096
#define D_OUT    4096
#define NPLANES  3
#define BM       128
#define BN       64
#define KB       32
#define KSTEPS   (D_IN / KB)
#define WEFF_SCALE 1024.0f

#if defined(__HIP_DEVICE_COMPILE__) && \
    __has_builtin(__builtin_amdgcn_global_load_async_to_lds_b128)
#define HAVE_ASYNC_LDS 1
#else
#define HAVE_ASYNC_LDS 0
#endif

// DAC: clamp to [-1,1], quantize to 127 signed levels, scale to volts.
__device__ __forceinline__ float dacf(float x) {
    float v = x * 0.15f;                       // INPUT_FACTOR
    v = fminf(fmaxf(v, -1.0f), 1.0f);
    v = rintf(v * 127.0f);                     // v_rndne_f32 == round-half-even
    return v * (0.6f / 127.0f);                // VMAX / levels
}

// ADC: current*OUT_SCALE, quantize step 2^-8, clip to +-16.
// Accumulator carries an extra WEFF_SCALE factor; fold its inverse in here.
__device__ __forceinline__ float adcf(float a) {
    float scaled = a * (8020.0f / WEFF_SCALE);
    float q = rintf(scaled * 256.0f) * (1.0f / 256.0f);
    return fminf(fmaxf(q, -16.0f), 16.0f);
}

union Frag { v16h h; v4u u[2]; };
union H4   { _Float16 h[4]; unsigned long long u64; };
union H8   { _Float16 h[8]; v4u u; };

#if HAVE_ASYNC_LDS
__device__ __forceinline__ void cp16_async(const _Float16* g, _Float16* l) {
    __builtin_amdgcn_global_load_async_to_lds_b128(
        (AS1 v4i_gcc*)(g),
        (AS3 v4i_gcc*)(l), 0, 0);
}
__device__ __forceinline__ void wait_async0() {
#if __has_builtin(__builtin_amdgcn_s_wait_asynccnt)
    __builtin_amdgcn_s_wait_asynccnt(0);
#else
    asm volatile("s_wait_asynccnt 0" ::: "memory");
#endif
}
#endif

__device__ __forceinline__ void cp16(const _Float16* g, _Float16* l) {
#if HAVE_ASYNC_LDS
    cp16_async(g, l);
#else
    *reinterpret_cast<v4u*>(l) = *reinterpret_cast<const v4u*>(g);
#endif
}

// ---------------------------------------------------------------------------
// Pre-pass 1: DAC-quantize x -> f16  (xh[b][i])
// ---------------------------------------------------------------------------
__global__ __launch_bounds__(256)
void dac_kernel(const float* __restrict__ x, _Float16* __restrict__ xh) {
    size_t i4 = (size_t)blockIdx.x * 256 + threadIdx.x;   // float4 index
    const float4 v = reinterpret_cast<const float4*>(x)[i4];
    H4 t;
    t.h[0] = (_Float16)dacf(v.x);
    t.h[1] = (_Float16)dacf(v.y);
    t.h[2] = (_Float16)dacf(v.z);
    t.h[3] = (_Float16)dacf(v.w);
    reinterpret_cast<unsigned long long*>(xh)[i4] = t.u64;
}

// ---------------------------------------------------------------------------
// Pre-pass 2: w_eff = (wpos - wneg)*WEFF_SCALE -> f16, transposed to
// wt[p][o][i] so the GEMM's B staging is contiguous along K.
// Block handles one 64(i) x 64(o) tile of one plane via LDS transpose.
// ---------------------------------------------------------------------------
__global__ __launch_bounds__(256)
void weff_kernel(const float* __restrict__ wpos, const float* __restrict__ wneg,
                 _Float16* __restrict__ wt) {
    __shared__ _Float16 sT[64][72];   // [i][o], padded to dodge bank conflicts
    const int tid = threadIdx.x;
    const int i0 = blockIdx.x * 64;
    const int o0 = blockIdx.y * 64;
    const int p  = blockIdx.z;

    #pragma unroll
    for (int j = 0; j < 4; ++j) {
        int idx = tid + j * 256;          // 0..1023
        int row = idx >> 4;               // i offset 0..63
        int ov  = idx & 15;               // float4 along o
        size_t g = ((size_t)p * D_IN + (i0 + row)) * D_OUT + o0 + ov * 4;
        float4 P = *reinterpret_cast<const float4*>(wpos + g);
        float4 N = *reinterpret_cast<const float4*>(wneg + g);
        sT[row][ov * 4 + 0] = (_Float16)((P.x - N.x) * WEFF_SCALE);
        sT[row][ov * 4 + 1] = (_Float16)((P.y - N.y) * WEFF_SCALE);
        sT[row][ov * 4 + 2] = (_Float16)((P.z - N.z) * WEFF_SCALE);
        sT[row][ov * 4 + 3] = (_Float16)((P.w - N.w) * WEFF_SCALE);
    }
    __syncthreads();
    #pragma unroll
    for (int j = 0; j < 2; ++j) {
        int idx = tid + j * 256;          // 0..511
        int o   = idx >> 3;               // o offset 0..63
        int seg = idx & 7;                // 8-half segment along i
        H8 t;
        #pragma unroll
        for (int e = 0; e < 8; ++e) t.h[e] = sT[seg * 8 + e][o];
        size_t g = ((size_t)p * D_OUT + (o0 + o)) * D_IN + i0 + seg * 8;
        *reinterpret_cast<v4u*>(wt + g) = t.u;
    }
}

// ---------------------------------------------------------------------------
// Main GEMM: xh[f16] @ wt[f16] per plane, WMMA f16->f32, ADC + shift-acc.
// Staging is pure 16B global->LDS copies (async on CDNA5), double-buffered.
// ---------------------------------------------------------------------------
__global__ __launch_bounds__(256)
void memristor_gemm_f16(const _Float16* __restrict__ xh,
                        const _Float16* __restrict__ wt,
                        const float* __restrict__ bias,
                        float* __restrict__ out) {
    __shared__ _Float16 sA[2][BM * KB];            // [buf][m][k]       8 KB/buf
    __shared__ _Float16 sB[2][NPLANES * BN * KB];  // [buf][p][n][k]   12 KB/buf

    const int tid  = threadIdx.x;
    const int lane = tid & 31;
    const int wid  = tid >> 5;
    const int wm   = wid & 3;     // 4 waves along tokens
    const int wn   = wid >> 2;    // 2 waves along outputs
    const int b0   = blockIdx.x * BM;   // token-major grid -> L2 weight reuse
    const int o0   = blockIdx.y * BN;

    // Per-thread copy slots (fixed across K steps)
    const int arow0 = (tid + 0 * 256) >> 2, aseg0 = (tid + 0 * 256) & 3;
    const int arow1 = (tid + 1 * 256) >> 2, aseg1 = (tid + 1 * 256) & 3;
    int bpl[3], bn[3], bsg[3];
    #pragma unroll
    for (int j = 0; j < 3; ++j) {
        int idx = tid + j * 256;          // 0..767
        bpl[j] = idx >> 8;                // plane
        bn[j]  = (idx & 255) >> 2;        // n 0..63
        bsg[j] = idx & 3;                 // 8-half segment along k
    }

    auto issue_copies = [&](int ks, int buf) {
        const int k0 = ks * KB;
        cp16(xh + (size_t)(b0 + arow0) * D_IN + k0 + aseg0 * 8,
             &sA[buf][arow0 * KB + aseg0 * 8]);
        cp16(xh + (size_t)(b0 + arow1) * D_IN + k0 + aseg1 * 8,
             &sA[buf][arow1 * KB + aseg1 * 8]);
        #pragma unroll
        for (int j = 0; j < 3; ++j) {
            cp16(wt + ((size_t)bpl[j] * D_OUT + (o0 + bn[j])) * D_IN + k0 + bsg[j] * 8,
                 &sB[buf][bpl[j] * (BN * KB) + bn[j] * KB + bsg[j] * 8]);
        }
        // CDNA5 global_prefetch_b8 one step further ahead on the weight stream
        if (ks + 1 < KSTEPS) {
            __builtin_prefetch(wt + ((size_t)bpl[0] * D_OUT + (o0 + bn[0])) * D_IN
                                  + (k0 + KB) + bsg[0] * 8, 0, 1);
        }
    };

    v8f acc[NPLANES][2][2];
    #pragma unroll
    for (int p = 0; p < NPLANES; ++p)
        #pragma unroll
        for (int i = 0; i < 2; ++i)
            #pragma unroll
            for (int j = 0; j < 2; ++j)
                #pragma unroll
                for (int e = 0; e < 8; ++e)
                    acc[p][i][j][e] = 0.0f;

    issue_copies(0, 0);
#if HAVE_ASYNC_LDS
    wait_async0();
#endif
    __syncthreads();

    const int arow = lane & 15;
    const int akb  = (lane < 16) ? 0 : 16;   // A: lanes 16-31 hold K=16..31
    const int bkb  = (lane < 16) ? 0 : 16;   // B: lanes 16-31 hold K=16..31

    for (int ks = 0; ks < KSTEPS; ++ks) {
        const int cur = ks & 1;
        if (ks + 1 < KSTEPS) issue_copies(ks + 1, cur ^ 1);

        Frag af[2];
        #pragma unroll
        for (int mt = 0; mt < 2; ++mt) {
            int m = wm * 32 + mt * 16 + arow;
            const _Float16* base = &sA[cur][m * KB];
            af[mt].u[0] = *reinterpret_cast<const v4u*>(base + akb);
            af[mt].u[1] = *reinterpret_cast<const v4u*>(base + 8 + akb);
        }
        #pragma unroll
        for (int p = 0; p < NPLANES; ++p) {
            #pragma unroll
            for (int nt = 0; nt < 2; ++nt) {
                int n = wn * 32 + nt * 16 + (lane & 15);
                const _Float16* base = &sB[cur][p * (BN * KB) + n * KB + bkb];
                Frag bf;
                bf.u[0] = *reinterpret_cast<const v4u*>(base);
                bf.u[1] = *reinterpret_cast<const v4u*>(base + 8);
                #pragma unroll
                for (int mt = 0; mt < 2; ++mt) {
                    acc[p][mt][nt] = __builtin_amdgcn_wmma_f32_16x16x32_f16(
                        false, af[mt].h, false, bf.h,
                        (short)0, acc[p][mt][nt], false, false);
                }
            }
        }
#if HAVE_ASYNC_LDS
        if (ks + 1 < KSTEPS) wait_async0();
#endif
        __syncthreads();
    }

    // ---- epilogue: ADC per plane, shift-accumulate [4,2,1], bias, store ----
    const int col   = lane & 15;
    const int rowhi = (lane >> 4) * 8;    // C layout: lanes 16-31 hold M = r + 8
    #pragma unroll
    for (int nt = 0; nt < 2; ++nt) {
        int og = o0 + wn * 32 + nt * 16 + col;
        float bv = bias[og];
        #pragma unroll
        for (int mt = 0; mt < 2; ++mt) {
            int bg0 = b0 + wm * 32 + mt * 16 + rowhi;
            #pragma unroll
            for (int r = 0; r < 8; ++r) {
                float q0 = adcf(acc[0][mt][nt][r]);
                float q1 = adcf(acc[1][mt][nt][r]);
                float q2 = adcf(acc[2][mt][nt][r]);
                float val = (4.0f * q0 + 2.0f * q1 + q2) * 0.01f + bv;
                out[(size_t)(bg0 + r) * D_OUT + og] = val;
            }
        }
    }
}

// ---------------------------------------------------------------------------
// Fallback: fused single-kernel version (round-1 style) when ws too small.
// ---------------------------------------------------------------------------
__global__ __launch_bounds__(256)
void memristor_fused_kernel(const float* __restrict__ x,
                            const float* __restrict__ wpos,
                            const float* __restrict__ wneg,
                            const float* __restrict__ bias,
                            float* __restrict__ out) {
    __shared__ _Float16 sA[2][BM * KB];
    __shared__ _Float16 sB[2][NPLANES * BN * KB];

    const int tid  = threadIdx.x;
    const int lane = tid & 31;
    const int wid  = tid >> 5;
    const int wm   = wid & 3;
    const int wn   = wid >> 2;
    const int b0   = blockIdx.x * BM;
    const int o0   = blockIdx.y * BN;

    v8f acc[NPLANES][2][2];
    #pragma unroll
    for (int p = 0; p < NPLANES; ++p)
        #pragma unroll
        for (int i = 0; i < 2; ++i)
            #pragma unroll
            for (int j = 0; j < 2; ++j)
                #pragma unroll
                for (int e = 0; e < 8; ++e)
                    acc[p][i][j][e] = 0.0f;

    float4 rA[4];
    float4 rP[6], rN[6];

    auto load_tile = [&](int ks) {
        const int k0 = ks * KB;
        #pragma unroll
        for (int j = 0; j < 4; ++j) {
            int idx = tid + j * 256;
            int row = idx >> 3, colv = idx & 7;
            rA[j] = *reinterpret_cast<const float4*>(
                x + (size_t)(b0 + row) * D_IN + k0 + colv * 4);
        }
        #pragma unroll
        for (int j = 0; j < 6; ++j) {
            int idx = tid + j * 256;
            int plane = idx >> 9, rem = idx & 511;
            int krow = rem >> 4, ov = rem & 15;
            size_t g = ((size_t)plane * D_IN + (k0 + krow)) * D_OUT + o0 + ov * 4;
            rP[j] = *reinterpret_cast<const float4*>(wpos + g);
            rN[j] = *reinterpret_cast<const float4*>(wneg + g);
        }
    };
    auto store_tile = [&](int buf) {
        #pragma unroll
        for (int j = 0; j < 4; ++j) {
            int idx = tid + j * 256;
            int row = idx >> 3, colv = idx & 7;
            H4 t;
            t.h[0] = (_Float16)dacf(rA[j].x);
            t.h[1] = (_Float16)dacf(rA[j].y);
            t.h[2] = (_Float16)dacf(rA[j].z);
            t.h[3] = (_Float16)dacf(rA[j].w);
            *reinterpret_cast<unsigned long long*>(&sA[buf][row * KB + colv * 4]) = t.u64;
        }
        #pragma unroll
        for (int j = 0; j < 6; ++j) {
            int idx = tid + j * 256;
            int plane = idx >> 9, rem = idx & 511;
            int krow = rem >> 4, ov = rem & 15;
            int nb = ov * 4;
            _Float16* bb = &sB[buf][plane * (BN * KB)];
            bb[(nb + 0) * KB + krow] = (_Float16)((rP[j].x - rN[j].x) * WEFF_SCALE);
            bb[(nb + 1) * KB + krow] = (_Float16)((rP[j].y - rN[j].y) * WEFF_SCALE);
            bb[(nb + 2) * KB + krow] = (_Float16)((rP[j].z - rN[j].z) * WEFF_SCALE);
            bb[(nb + 3) * KB + krow] = (_Float16)((rP[j].w - rN[j].w) * WEFF_SCALE);
        }
    };

    load_tile(0);
    store_tile(0);
    __syncthreads();

    const int arow = lane & 15;
    const int akb  = (lane < 16) ? 0 : 16;
    const int bkb  = (lane < 16) ? 0 : 16;

    for (int ks = 0; ks < KSTEPS; ++ks) {
        const int cur = ks & 1;
        if (ks + 1 < KSTEPS) load_tile(ks + 1);

        Frag af[2];
        #pragma unroll
        for (int mt = 0; mt < 2; ++mt) {
            int m = wm * 32 + mt * 16 + arow;
            const _Float16* base = &sA[cur][m * KB];
            af[mt].u[0] = *reinterpret_cast<const v4u*>(base + akb);
            af[mt].u[1] = *reinterpret_cast<const v4u*>(base + 8 + akb);
        }
        #pragma unroll
        for (int p = 0; p < NPLANES; ++p) {
            #pragma unroll
            for (int nt = 0; nt < 2; ++nt) {
                int n = wn * 32 + nt * 16 + (lane & 15);
                const _Float16* base = &sB[cur][p * (BN * KB) + n * KB + bkb];
                Frag bf;
                bf.u[0] = *reinterpret_cast<const v4u*>(base);
                bf.u[1] = *reinterpret_cast<const v4u*>(base + 8);
                #pragma unroll
                for (int mt = 0; mt < 2; ++mt) {
                    acc[p][mt][nt] = __builtin_amdgcn_wmma_f32_16x16x32_f16(
                        false, af[mt].h, false, bf.h,
                        (short)0, acc[p][mt][nt], false, false);
                }
            }
        }
        __syncthreads();
        if (ks + 1 < KSTEPS) store_tile((ks + 1) & 1);
        __syncthreads();
    }

    const int col   = lane & 15;
    const int rowhi = (lane >> 4) * 8;
    #pragma unroll
    for (int nt = 0; nt < 2; ++nt) {
        int og = o0 + wn * 32 + nt * 16 + col;
        float bv = bias[og];
        #pragma unroll
        for (int mt = 0; mt < 2; ++mt) {
            int bg0 = b0 + wm * 32 + mt * 16 + rowhi;
            #pragma unroll
            for (int r = 0; r < 8; ++r) {
                float q0 = adcf(acc[0][mt][nt][r]);
                float q1 = adcf(acc[1][mt][nt][r]);
                float q2 = adcf(acc[2][mt][nt][r]);
                float val = (4.0f * q0 + 2.0f * q1 + q2) * 0.01f + bv;
                out[(size_t)(bg0 + r) * D_OUT + og] = val;
            }
        }
    }
}

extern "C" void kernel_launch(void* const* d_in, const int* in_sizes, int n_in,
                              void* d_out, int out_size, void* d_ws, size_t ws_size,
                              hipStream_t stream) {
    (void)in_sizes; (void)n_in; (void)out_size;
    const float* x    = (const float*)d_in[0];
    const float* wpos = (const float*)d_in[1];
    const float* wneg = (const float*)d_in[2];
    const float* bias = (const float*)d_in[3];
    float* out = (float*)d_out;

    const size_t xh_bytes = (size_t)TOKENS * D_IN * sizeof(_Float16);            // 64 MB
    const size_t wt_bytes = (size_t)NPLANES * D_OUT * D_IN * sizeof(_Float16);   // 96 MB

    if (ws_size >= xh_bytes + wt_bytes) {
        _Float16* xh = (_Float16*)d_ws;
        _Float16* wt = (_Float16*)((char*)d_ws + xh_bytes);

        dac_kernel<<<dim3((TOKENS * (size_t)D_IN) / 4 / 256), dim3(256), 0, stream>>>(x, xh);
        weff_kernel<<<dim3(D_IN / 64, D_OUT / 64, NPLANES), dim3(256), 0, stream>>>(
            wpos, wneg, wt);
        memristor_gemm_f16<<<dim3(TOKENS / BM, D_OUT / BN), dim3(256), 0, stream>>>(
            xh, wt, bias, out);
    } else {
        memristor_fused_kernel<<<dim3(TOKENS / BM, D_OUT / BN), dim3(256), 0, stream>>>(
            x, wpos, wneg, bias, out);
    }
}
